// ThreeLayerGATWithEdgeFeatures_29094108463689
// MI455X (gfx1250) — matrix operand order repaired and checked by reading, not compiled, not run on hardware
//
#include <hip/hip_runtime.h>

typedef __attribute__((ext_vector_type(16))) _Float16 v16h;
typedef __attribute__((ext_vector_type(8)))  float    v8f;

#define NNODES 50000
#define NEDGES 800000
#define HD     64           // HID (e_feat width, We rows)
#define NEG_SLOPE 0.2f

// ---------- helpers: order-preserving float<->uint for atomicMax ----------
__device__ __forceinline__ unsigned enc_f32(float f) {
  unsigned u = __float_as_uint(f);
  return (u & 0x80000000u) ? ~u : (u | 0x80000000u);
}
__device__ __forceinline__ float dec_u32(unsigned u) {
  u = (u & 0x80000000u) ? (u & 0x7FFFFFFFu) : ~u;
  return __uint_as_float(u);
}

// ---------- fill kernels (workspace must be re-inited every call) ----------
__global__ void fill_f32_kernel(float* p, float v, long long n) {
  long long t = (long long)blockIdx.x * blockDim.x + threadIdx.x;
  if (t < n) p[t] = v;
}
__global__ void fill_u32_kernel(unsigned* p, unsigned v, long long n) {
  long long t = (long long)blockIdx.x * blockDim.x + threadIdx.x;
  if (t < n) p[t] = v;
}

// ---------- WMMA GEMM: C[M,Nc] = act(A[M,KT] @ B[KT,Nc] [+ bias]) ----------
// fp32 in/out, fp16 multiply via v_wmma_f32_16x16x32_f16.
// One wave per 16x16 tile, 8 waves per block. KT/RELU/BIAS compile-time so
// all K bound checks and the epilogue fold to straight-line code; A fragment
// loaded as 4x float4 (global_load_b128).
template<int KT, bool RELU, bool BIAS>
__global__ void __launch_bounds__(256)
wmma_gemm_kernel(const float* __restrict__ A, const float* __restrict__ B,
                 const float* __restrict__ bias, float* __restrict__ C,
                 int M, int Nc) {
  const int lane   = threadIdx.x & 31;
  const int wave   = threadIdx.x >> 5;
  const int tilesN = Nc >> 4;
  const long long tile = (long long)blockIdx.x * 8 + wave;
  if (tile >= (long long)(M >> 4) * tilesN) return;
  const int tM   = (int)(tile / tilesN);
  const int tN   = (int)(tile % tilesN);
  const int half = lane >> 4;             // 0: lanes 0-15, 1: lanes 16-31
  const int mr   = tM * 16 + (lane & 15); // A row handled by this lane
  const int nc   = tN * 16 + (lane & 15); // B/C column handled by this lane
  const float* __restrict__ Arow = A + (size_t)mr * KT;

  v8f acc = {};
#pragma unroll
  for (int k0 = 0; k0 < KT; k0 += 32) {
    if (k0 + 32 < KT)  // static after unroll -> global_prefetch_b8
      __builtin_prefetch(Arow + k0 + 32, 0, 3);

    // A fragment (16x32 MxK): lanes 0-15 own K k0+[0..7]&[16..23],
    //                         lanes 16-31 own K k0+[8..15]&[24..31]
    float af[16];
    {
      const float4 p0 = *(const float4*)(Arow + k0 + half * 8);
      const float4 p1 = *(const float4*)(Arow + k0 + half * 8 + 4);
      af[0] = p0.x; af[1] = p0.y; af[2] = p0.z; af[3] = p0.w;
      af[4] = p1.x; af[5] = p1.y; af[6] = p1.z; af[7] = p1.w;
    }
    if (k0 + 16 < KT) {  // static after unroll (false only for KT==16)
      const float4 p2 = *(const float4*)(Arow + k0 + 16 + half * 8);
      const float4 p3 = *(const float4*)(Arow + k0 + 16 + half * 8 + 4);
      af[8]  = p2.x; af[9]  = p2.y; af[10] = p2.z; af[11] = p2.w;
      af[12] = p3.x; af[13] = p3.y; af[14] = p3.z; af[15] = p3.w;
    } else {
#pragma unroll
      for (int i = 8; i < 16; ++i) af[i] = 0.f;
    }
    v16h a, b;
#pragma unroll
    for (int i = 0; i < 16; ++i) a[i] = (_Float16)af[i];

    // B fragment (32x16 KxN): lanes 0-15 own K k0+[0..7]&[16..23],
    //                         lanes 16-31 own K k0+[8..15]&[24..31]
    {
      const float* __restrict__ Bp = B + (size_t)(k0 + half * 8) * Nc + nc;
#pragma unroll
      for (int i = 0; i < 16; ++i) {
        const int kr = (i < 8) ? i : i + 8;
        b[i] = (k0 + half * 8 + kr < KT) ? (_Float16)Bp[(size_t)kr * Nc] : (_Float16)0.f;
      }
    }

    acc = __builtin_amdgcn_wmma_f32_16x16x32_f16(false, a, false, b,
                                                 (short)0, acc, false, false);
  }

  float bv = 0.f;
  if (BIAS) bv = bias[nc];
  float* __restrict__ Cp = C + (size_t)(tM * 16 + 8 * half) * Nc + nc;
#pragma unroll
  for (int v = 0; v < 8; ++v) {       // D layout: VGPR v -> M = v + 8*half
    float val = acc[v];
    if (BIAS) val += bv;
    if (RELU) val = fmaxf(val, 0.f);
    Cp[(size_t)v * Nc] = val;
  }
}

// ---------- a_src / a_dst: per-(node,head) dot of xp row with att vectors ----
template<int H, int C>
__global__ void node_att_kernel(const float* __restrict__ xp,
                                const float* __restrict__ att_s,
                                const float* __restrict__ att_d,
                                float* __restrict__ a_src, float* __restrict__ a_dst,
                                int Nn) {
  int t = blockIdx.x * blockDim.x + threadIdx.x;
  if (t >= Nn * H) return;
  int n = t / H, h = t % H;                 // H power of two -> shift/mask
  const float* row = xp + (size_t)n * (H * C) + h * C;
  float ss = 0.f, sd = 0.f;
#pragma unroll 4
  for (int c = 0; c < C; ++c) {
    float v = row[c];
    ss += v * att_s[h * C + c];
    sd += v * att_d[h * C + c];
  }
  a_src[t] = ss;
  a_dst[t] = sd;
}

// ---------- fold We with att_e:  we_att[k,h] = sum_c We[k, h*C+c]*att_e[h,c] --
template<int H, int C>
__global__ void fold_we_kernel(const float* __restrict__ We,
                               const float* __restrict__ att_e,
                               float* __restrict__ we_att) {
  int t = blockIdx.x * blockDim.x + threadIdx.x;
  if (t >= HD * H) return;
  int k = t / H, h = t % H;
  float s = 0.f;
  for (int c = 0; c < C; ++c) s += We[(size_t)k * (H * C) + h * C + c] * att_e[h * C + c];
  we_att[t] = s;
}

// ---------- a_edge[e,h] = e_feat[e,:] @ we_att[:,h] --------------------------
template<int H>
__global__ void edge_score_kernel(const float* __restrict__ e_feat,
                                  const float* __restrict__ we_att,
                                  float* __restrict__ a_edge, long long E_) {
  long long t = (long long)blockIdx.x * blockDim.x + threadIdx.x;
  if (t >= E_ * H) return;
  long long e = t / H; int h = (int)(t % H);
  const float* row = e_feat + e * HD;
  float s = 0.f;
#pragma unroll 4
  for (int k = 0; k < HD; ++k) s += row[k] * we_att[k * H + h];
  a_edge[t] = s;
}

// ---------- pass 1: alpha = leaky_relu(...), atomicMax into amax -------------
template<int H>
__global__ void edge_alpha_kernel(const int* __restrict__ src, const int* __restrict__ dst,
                                  const float* __restrict__ a_src, const float* __restrict__ a_dst,
                                  const float* __restrict__ a_edge,
                                  float* __restrict__ alpha, unsigned* __restrict__ amax,
                                  long long E_) {
  long long t = (long long)blockIdx.x * blockDim.x + threadIdx.x;
  if (t >= E_ * H) return;
  long long e = t / H; int h = (int)(t % H);
  int s = src[e], d = dst[e];
  float a = a_src[s * H + h] + a_dst[d * H + h] + a_edge[t];
  a = (a > 0.f) ? a : NEG_SLOPE * a;
  alpha[t] = a;
  atomicMax(&amax[d * H + h], enc_f32(a));
}

// ---------- pass 2: ex = exp(alpha - amax[dst]); atomicAdd denom -------------
template<int H>
__global__ void edge_exp_kernel(const int* __restrict__ dst,
                                const unsigned* __restrict__ amax,
                                float* __restrict__ alpha, float* __restrict__ denom,
                                long long E_) {
  long long t = (long long)blockIdx.x * blockDim.x + threadIdx.x;
  if (t >= E_ * H) return;
  long long e = t / H; int h = (int)(t % H);
  int d = dst[e];
  float ex = __expf(alpha[t] - dec_u32(amax[d * H + h]));
  alpha[t] = ex;
  atomicAdd(&denom[d * H + h], ex);
}

// ---------- pass 3: agg[dst] += xp[src] * (ex/denom[dst]) --------------------
// blockIdx carries the edge index -> zero integer divisions.
template<int H, int C>
__global__ void __launch_bounds__(256)
edge_msg_kernel(const int* __restrict__ src, const int* __restrict__ dst,
                const float* __restrict__ ex, const float* __restrict__ denom,
                const float* __restrict__ xp, float* __restrict__ agg, long long E_) {
  constexpr int HC  = H * C;
  constexpr int EPB = 256 / HC;           // edges per block
  long long e = (long long)blockIdx.x * EPB + (threadIdx.x / HC);
  int j = threadIdx.x % HC;               // power-of-two mask
  if (e >= E_) return;
  int h = j / C;
  int s = src[e], d = dst[e];
  float w = ex[e * H + h] / (denom[d * H + h] + 1e-16f);
  atomicAdd(&agg[(size_t)d * HC + j], xp[(size_t)s * HC + j] * w);
}

// ---------- finalize: out = act(agg + bias) ----------------------------------
template<int COLS, bool RELU>
__global__ void finalize_kernel(float* __restrict__ agg, const float* __restrict__ bias,
                                long long n) {
  long long t = (long long)blockIdx.x * blockDim.x + threadIdx.x;
  if (t >= n) return;
  int j = (int)(t % COLS);                // power-of-two mask
  float v = agg[t] + bias[j];
  if (RELU) v = fmaxf(v, 0.f);
  agg[t] = v;
}

// ================= host side =================================================
static unsigned cdiv_u(long long a, long long b) { return (unsigned)((a + b - 1) / b); }

template<int KT, bool RELU, bool BIAS>
static void launch_gemm(const float* A, const float* Bw, const float* bias, float* C,
                        int M, int Nc, hipStream_t stream) {
  long long tiles = (long long)(M >> 4) * (Nc >> 4);
  unsigned g = cdiv_u(tiles, 8);
  wmma_gemm_kernel<KT, RELU, BIAS><<<g, 256, 0, stream>>>(A, Bw, bias, C, M, Nc);
}

template<int H, int C, int KT, bool RELU_OUT>
static void run_gat(const float* h_in, const float* W, const float* as_,
                    const float* ad_, const float* We_, const float* ae_, const float* bb,
                    float* xp, float* outp,
                    const int* src, const int* dst, const float* e_feat,
                    float* a_src_, float* a_dst_, float* a_edge, float* alpha,
                    unsigned* amax, float* denom_, float* we_att, hipStream_t stream) {
  const int Nn = NNODES;
  const long long Ee = NEDGES;
  constexpr int HC = H * C;
  constexpr int EPB = 256 / HC;

  launch_gemm<KT, false, false>(h_in, W, nullptr, xp, Nn, HC, stream);   // xp = h @ W
  node_att_kernel<H, C><<<cdiv_u((long long)Nn * H, 256), 256, 0, stream>>>(
      xp, as_, ad_, a_src_, a_dst_, Nn);
  fold_we_kernel<H, C><<<cdiv_u(HD * H, 64), 64, 0, stream>>>(We_, ae_, we_att);
  edge_score_kernel<H><<<cdiv_u(Ee * H, 256), 256, 0, stream>>>(e_feat, we_att, a_edge, Ee);
  fill_u32_kernel<<<cdiv_u((long long)Nn * H, 256), 256, 0, stream>>>(amax, 0u, (long long)Nn * H);
  fill_f32_kernel<<<cdiv_u((long long)Nn * H, 256), 256, 0, stream>>>(denom_, 0.f, (long long)Nn * H);
  fill_f32_kernel<<<cdiv_u((long long)Nn * HC, 256), 256, 0, stream>>>(outp, 0.f, (long long)Nn * HC);
  edge_alpha_kernel<H><<<cdiv_u(Ee * H, 256), 256, 0, stream>>>(
      src, dst, a_src_, a_dst_, a_edge, alpha, amax, Ee);
  edge_exp_kernel<H><<<cdiv_u(Ee * H, 256), 256, 0, stream>>>(dst, amax, alpha, denom_, Ee);
  edge_msg_kernel<H, C><<<cdiv_u(Ee, EPB), 256, 0, stream>>>(
      src, dst, alpha, denom_, xp, outp, Ee);
  finalize_kernel<HC, RELU_OUT><<<cdiv_u((long long)Nn * HC, 256), 256, 0, stream>>>(
      outp, bb, (long long)Nn * HC);
}

extern "C" void kernel_launch(void* const* d_in, const int* in_sizes, int n_in,
                              void* d_out, int out_size, void* d_ws, size_t ws_size,
                              hipStream_t stream) {
  const int Nn = NNODES;
  const long long Ee = NEDGES;

  const float* x         = (const float*)d_in[0];
  const int*   eidx      = (const int*)d_in[1];
  const float* edge_attr = (const float*)d_in[2];
  const float* enc_W  = (const float*)d_in[3];
  const float* enc_b  = (const float*)d_in[4];
  const float* proj_W = (const float*)d_in[5];
  const float* proj_b = (const float*)d_in[6];
  const float* W1 = (const float*)d_in[7],  *as1 = (const float*)d_in[8],
             *ad1 = (const float*)d_in[9],  *We1 = (const float*)d_in[10],
             *ae1 = (const float*)d_in[11], *b1  = (const float*)d_in[12];
  const float* W2 = (const float*)d_in[13], *as2 = (const float*)d_in[14],
             *ad2 = (const float*)d_in[15], *We2 = (const float*)d_in[16],
             *ae2 = (const float*)d_in[17], *b2  = (const float*)d_in[18];
  const float* W3 = (const float*)d_in[19], *as3 = (const float*)d_in[20],
             *ad3 = (const float*)d_in[21], *We3 = (const float*)d_in[22],
             *ae3 = (const float*)d_in[23], *b3  = (const float*)d_in[24];
  const int* src = eidx;
  const int* dst = eidx + Ee;
  float* out = (float*)d_out;

  // ---- carve workspace ----
  char* wp = (char*)d_ws;
  auto carve = [&](size_t bytes) -> void* {
    void* r = (void*)wp;
    wp += (bytes + 255) & ~(size_t)255;
    return r;
  };
  float*    e_feat = (float*)   carve((size_t)Ee * HD * 4);   // [E,64]
  float*    B0     = (float*)   carve((size_t)Nn * 256 * 4);  // rotating node buffers
  float*    B1     = (float*)   carve((size_t)Nn * 256 * 4);
  float*    B2     = (float*)   carve((size_t)Nn * 256 * 4);
  float*    a_src_ = (float*)   carve((size_t)Nn * 4 * 4);
  float*    a_dst_ = (float*)   carve((size_t)Nn * 4 * 4);
  float*    a_edge = (float*)   carve((size_t)Ee * 4 * 4);
  float*    alpha  = (float*)   carve((size_t)Ee * 4 * 4);    // reused as ex
  unsigned* amax   = (unsigned*)carve((size_t)Nn * 4 * 4);
  float*    denom_ = (float*)   carve((size_t)Nn * 4 * 4);
  float*    we_att = (float*)   carve((size_t)HD * 4 * 4);

  // ---- shared front end ----
  launch_gemm<16, true, true>(edge_attr, enc_W, enc_b, e_feat, (int)Ee, HD, stream); // e_feat
  launch_gemm<32, true, true>(x, proj_W, proj_b, B0, Nn, HD, stream);                // h

  // layer 1: in=64,  heads=4, C=64, concat, relu   (h: B0 -> xp: B1 -> out: B2)
  run_gat<4, 64, 64, true>(B0, W1, as1, ad1, We1, ae1, b1, B1, B2,
                           src, dst, e_feat, a_src_, a_dst_, a_edge, alpha,
                           amax, denom_, we_att, stream);
  // layer 2: in=256, heads=4, C=64, concat, relu   (h: B2 -> xp: B0 -> out: B1)
  run_gat<4, 64, 256, true>(B2, W2, as2, ad2, We2, ae2, b2, B0, B1,
                            src, dst, e_feat, a_src_, a_dst_, a_edge, alpha,
                            amax, denom_, we_att, stream);
  // layer 3: in=256, heads=1, C=32, mean (=identity for H=1), no relu -> d_out
  run_gat<1, 32, 256, false>(B1, W3, as3, ad3, We3, ae3, b3, B2, out,
                             src, dst, e_feat, a_src_, a_dst_, a_edge, alpha,
                             amax, denom_, we_att, stream);
}
